// Net3_19791209300083
// MI455X (gfx1250) — matrix operand order repaired and checked by reading, not compiled, not run on hardware
//
#include <hip/hip_runtime.h>
#include <math.h>

typedef __attribute__((ext_vector_type(16))) _Float16 v16h;
typedef __attribute__((ext_vector_type(8)))  float    v8f;

#define DIVUP(a, b) (((a) + (b) - 1) / (b))

// ---------------------------------------------------------------------------
// Float atomic max via monotonic int mapping
// ---------------------------------------------------------------------------
__device__ __forceinline__ void atomicMaxF32(float* addr, float val) {
  if (val >= 0.0f) {
    atomicMax((int*)addr, __float_as_int(val));
  } else {
    atomicMin((unsigned int*)addr, (unsigned int)__float_as_int(val));
  }
}

// ---------------------------------------------------------------------------
// Prep kernels
// ---------------------------------------------------------------------------
__global__ void k_fill_f32(float* p, float v, int n) {
  int t = blockIdx.x * blockDim.x + threadIdx.x;
  if (t < n) p[t] = v;
}

// Pack activations f32 [n, fin] into WMMA A-fragment layout (f16):
//   Apk[((tm*2 + kb)*32 + lane)*16 + h],
//   lane = m%16 + 16*((kk/8)&1), h = kk%8 + 8*(kk>=16), kk = k%32, kb = k/32
// (pad region pre-zeroed by memset when fin < 32)
__global__ void k_act_pack(const float* __restrict__ X, _Float16* __restrict__ Apk,
                           int n, int fin) {
  int t = blockIdx.x * blockDim.x + threadIdx.x;
  if (t >= n * fin) return;
  int r = t / fin, k = t - r * fin;
  int tm = r >> 4, m16 = r & 15;
  int kb = k >> 5, kk = k & 31;
  int lane = m16 + 16 * ((kk >> 3) & 1);
  int h = (kk & 7) + ((kk & 16) ? 8 : 0);
  Apk[((((size_t)tm * 2 + kb) * 32 + lane) << 4) + h] = (_Float16)X[t];
}

// Pack weight f32 [fin, fout] into WMMA B-fragment layout (f16) at col colOff:
//   Wpk[((tn*KB + kb)*32 + lane)*16 + h], lane = n%16 + 16*((kk/8)&1), h = kk%8 + 8*(kk>=16)
__global__ void k_conv_w_pack(const float* __restrict__ W, _Float16* __restrict__ Wpk,
                              int fin, int fout, int KB, int colOff) {
  int t = blockIdx.x * blockDim.x + threadIdx.x;
  if (t >= fin * fout) return;
  int k = t / fout, c = t - k * fout;
  int n = colOff + c;
  int tn = n >> 4, l16 = n & 15;
  int kb = k >> 5, kk = k & 31;
  int lane = l16 + 16 * ((kk >> 3) & 1);
  int h = (kk & 7) + ((kk & 16) ? 8 : 0);
  Wpk[((((size_t)tn * KB + kb) * 32 + lane) << 4) + h] = (_Float16)W[t];
}

__global__ void k_copy_bias(const float* __restrict__ b, float* __restrict__ Bc,
                            int n, int off) {
  int t = blockIdx.x * blockDim.x + threadIdx.x;
  if (t < n) Bc[off + t] = b[t];
}

// ---------------------------------------------------------------------------
// WMMA GEMM on pre-packed fragments:
//   C[M, ldw] = A[M, 32*KB] x W + bias, one wave per 16x16 tile.
//   Inner loop: 2x 32B vector loads + 1x v_wmma_f32_16x16x32_f16.
//   A pack stride is always 2 k-blocks (64-wide activation buffer).
// ---------------------------------------------------------------------------
__global__ void __launch_bounds__(256)
k_gemm_wmma(const _Float16* __restrict__ Apk, const _Float16* __restrict__ Wpk,
            const float* __restrict__ bias, float* __restrict__ Cout,
            int M, int KB, int ldw, int tilesN) {
  int wave = blockIdx.x * (blockDim.x >> 5) + (threadIdx.x >> 5);
  int lane = threadIdx.x & 31;
  int tilesM = (M + 15) >> 4;
  if (wave >= tilesM * tilesN) return;
  int tm = wave / tilesN, tn = wave - tm * tilesN;

  const _Float16* ap = Apk + ((((size_t)tm * 2) * 32 + lane) << 4);
  const _Float16* bp = Wpk + ((((size_t)tn * KB) * 32 + lane) << 4);

  v8f acc = {};
  for (int kb = 0; kb < KB; ++kb) {
    v16h a = *(const v16h*)(ap + (size_t)kb * 512);
    v16h b = *(const v16h*)(bp + (size_t)kb * 512);
    acc = __builtin_amdgcn_wmma_f32_16x16x32_f16(
        /*neg_a=*/false, a, /*neg_b=*/false, b,
        /*c_mod=*/(short)0, acc, /*reuse_a=*/false, /*reuse_b=*/false);
  }

  int lane16 = lane & 15, laneGroup = lane >> 4;
  int colB = (tn << 4) + lane16;
  float bb = bias[colB];
  int row0 = (tm << 4) + 8 * laneGroup;
  float* cp = Cout + (size_t)row0 * ldw + colB;
  if (((tm + 1) << 4) <= M) {       // wave-uniform fast path: no per-row guards
#pragma unroll
    for (int r = 0; r < 8; ++r) cp[(size_t)r * ldw] = acc[r] + bb;
  } else {
    for (int r = 0; r < 8; ++r)
      if (row0 + r < M) cp[(size_t)r * ldw] = acc[r] + bb;
  }
}

// ---------------------------------------------------------------------------
// Edge pass 1 (wave per (edge,head)): lanes stripe channels, wave32 shfl
// reduction, lane 0 writes logit + segment max via atomic.
// ---------------------------------------------------------------------------
__global__ void __launch_bounds__(256)
k_edge_logits(const float* __restrict__ Z, const int* __restrict__ src,
              const int* __restrict__ dst, float* __restrict__ logits,
              float* __restrict__ m, int E, int H, int C, int stride,
              int qoff, int koff, float scale) {
  int wid = blockIdx.x * (blockDim.x >> 5) + (threadIdx.x >> 5);
  int lane = threadIdx.x & 31;
  if (wid >= E * H) return;
  int e = wid / H, h = wid - e * H;
  int s = src[e], d = dst[e];
  const float* q = Z + (size_t)d * stride + qoff + h * C;
  const float* k = Z + (size_t)s * stride + koff + h * C;
  float acc = 0.0f;
  for (int c = lane; c < C; c += 32) acc += q[c] * k[c];
#pragma unroll
  for (int o = 16; o > 0; o >>= 1) acc += __shfl_xor(acc, o, 32);
  if (lane == 0) {
    float lg = acc * scale;
    logits[wid] = lg;
    atomicMaxF32(&m[d * H + h], lg);
  }
}

// ---------------------------------------------------------------------------
// Edge pass 2 (thread per (edge,head)): e = exp(logit - m); denom += e
// ---------------------------------------------------------------------------
__global__ void k_edge_exp(float* __restrict__ logits, const int* __restrict__ dst,
                           const float* __restrict__ m, float* __restrict__ denom,
                           int E, int H) {
  int t = blockIdx.x * blockDim.x + threadIdx.x;
  if (t >= E * H) return;
  int e = t / H, h = t - e * H;
  int d = dst[e];
  float ev = __expf(logits[t] - m[d * H + h]);
  logits[t] = ev;
  atomicAdd(&denom[d * H + h], ev);
}

// ---------------------------------------------------------------------------
// Edge pass 3 (wave per (edge,head)): lanes stripe channels; 32-wide
// parallel scatter-add of alpha * v[src] into agg[dst].
// ---------------------------------------------------------------------------
__global__ void __launch_bounds__(256)
k_edge_agg(const float* __restrict__ Z, const int* __restrict__ src,
           const int* __restrict__ dst, const float* __restrict__ logits,
           const float* __restrict__ denom, float* __restrict__ agg,
           int E, int H, int C, int stride, int voff) {
  int wid = blockIdx.x * (blockDim.x >> 5) + (threadIdx.x >> 5);
  int lane = threadIdx.x & 31;
  if (wid >= E * H) return;
  int e = wid / H, h = wid - e * H;
  int s = src[e], d = dst[e];
  float alpha = logits[wid] / (denom[d * H + h] + 1e-16f);
  const float* v = Z + (size_t)s * stride + voff + h * C;
  float* ag = agg + ((size_t)d * H + h) * C;
  for (int c = lane; c < C; c += 32) atomicAdd(&ag[c], alpha * v[c]);
}

// ---------------------------------------------------------------------------
// Node epilogue, layers 1/2: out = relu(mean_h(agg) + skip) [+ hprev residual]
// ---------------------------------------------------------------------------
__global__ void k_out12(const float* __restrict__ agg, const float* __restrict__ Z,
                        int stride, int skipoff, const float* __restrict__ hprev,
                        float* __restrict__ out, int N, int H, int C) {
  int t = blockIdx.x * blockDim.x + threadIdx.x;
  if (t >= N * C) return;
  int n = t / C, c = t - n * C;
  float s = 0.0f;
  for (int h = 0; h < H; ++h) s += agg[((size_t)n * H + h) * C + c];
  float val = s * (1.0f / (float)H) + Z[(size_t)n * stride + skipoff + c];
  val = fmaxf(val, 0.0f);
  if (hprev) val += hprev[t];
  out[t] = val;
}

// ---------------------------------------------------------------------------
// Node epilogue, layer 3: out = log_softmax(agg + skip) over 10 classes
// ---------------------------------------------------------------------------
__global__ void k_out3(const float* __restrict__ agg, const float* __restrict__ Z,
                       int stride, int skipoff, float* __restrict__ out, int N) {
  int n = blockIdx.x * blockDim.x + threadIdx.x;
  if (n >= N) return;
  float v[10];
  float mx = -INFINITY;
#pragma unroll
  for (int c = 0; c < 10; ++c) {
    float t = agg[(size_t)n * 10 + c] + Z[(size_t)n * stride + skipoff + c];
    v[c] = t;
    mx = fmaxf(mx, t);
  }
  float s = 0.0f;
#pragma unroll
  for (int c = 0; c < 10; ++c) s += __expf(v[c] - mx);
  float lse = mx + __logf(s);
#pragma unroll
  for (int c = 0; c < 10; ++c) out[(size_t)n * 10 + c] = v[c] - lse;
}

// ---------------------------------------------------------------------------
// Host
// ---------------------------------------------------------------------------
extern "C" void kernel_launch(void* const* d_in, const int* in_sizes, int n_in,
                              void* d_out, int out_size, void* d_ws, size_t ws_size,
                              hipStream_t stream) {
  (void)n_in; (void)out_size; (void)ws_size;
  const int N = in_sizes[0] / 13;
  const int E = in_sizes[1] / 2;
  const int tilesM = DIVUP(N, 16);
  const float* x = (const float*)d_in[0];
  const int* ei = (const int*)d_in[1];
  const int* src = ei;           // edge_index[0, :]
  const int* dst = ei + E;       // edge_index[1, :]

  // ---- scratch carve-out -------------------------------------------------
  size_t off = 0;
  auto carve = [&](size_t bytes) -> char* {
    char* p = (char*)d_ws + off;
    off = (off + bytes + 255) & ~(size_t)255;
    return p;
  };
  float*    Z      = (float*)carve((size_t)N * 640 * 4);        // Q|K|V|S
  _Float16* Apk    = (_Float16*)carve((size_t)tilesM * 1024 * 2); // packed A frags
  float*    h1     = (float*)carve((size_t)N * 64 * 4);
  float*    h2     = (float*)carve((size_t)N * 64 * 4);
  float*    logits = (float*)carve((size_t)E * 3 * 4);
  float*    m      = (float*)carve((size_t)N * 3 * 4);
  float*    denom  = (float*)carve((size_t)N * 3 * 4);
  float*    agg    = (float*)carve((size_t)N * 192 * 4);
  _Float16* Wpk1   = (_Float16*)carve((size_t)40 * 1 * 512 * 2); // KB=1
  _Float16* Wpk2   = (_Float16*)carve((size_t)40 * 2 * 512 * 2); // KB=2
  _Float16* Wpk3   = (_Float16*)carve((size_t)4 * 2 * 512 * 2);  // KB=2
  float*    Bc1    = (float*)carve(640 * 4);
  float*    Bc2    = (float*)carve(640 * 4);
  float*    Bc3    = (float*)carve(64 * 4);

  // ---- prep: zero pads, pack weights/biases ------------------------------
  hipMemsetAsync(Apk,  0, (size_t)tilesM * 1024 * 2, stream);   // layer-1 K pad
  hipMemsetAsync(Wpk1, 0, (size_t)40 * 512 * 2, stream);        // k >= 13 pad
  hipMemsetAsync(Wpk3, 0, (size_t)4 * 2 * 512 * 2, stream);     // col pad
  hipMemsetAsync(Bc3,  0, 64 * 4, stream);

  auto convW = [&](int idx, _Float16* Wpk, int fin, int fout, int KB, int colOff,
                   float* Bc) {
    const float* W = (const float*)d_in[idx];
    const float* b = (const float*)d_in[idx + 1];
    int n = fin * fout;
    k_conv_w_pack<<<DIVUP(n, 256), 256, 0, stream>>>(W, Wpk, fin, fout, KB, colOff);
    k_copy_bias<<<DIVUP(fout, 256), 256, 0, stream>>>(b, Bc, fout, colOff);
  };
  convW(2,  Wpk1, 13, 192, 1, 0,   Bc1);  // q1
  convW(4,  Wpk1, 13, 192, 1, 192, Bc1);  // k1
  convW(6,  Wpk1, 13, 192, 1, 384, Bc1);  // v1
  convW(8,  Wpk1, 13, 64,  1, 576, Bc1);  // s1
  convW(10, Wpk2, 64, 192, 2, 0,   Bc2);  // q2
  convW(12, Wpk2, 64, 192, 2, 192, Bc2);  // k2
  convW(14, Wpk2, 64, 192, 2, 384, Bc2);  // v2
  convW(16, Wpk2, 64, 64,  2, 576, Bc2);  // s2
  convW(18, Wpk3, 64, 10,  2, 0,   Bc3);  // q3
  convW(20, Wpk3, 64, 10,  2, 16,  Bc3);  // k3
  convW(22, Wpk3, 64, 10,  2, 32,  Bc3);  // v3
  convW(24, Wpk3, 64, 10,  2, 48,  Bc3);  // s3

  auto run_gemm = [&](const _Float16* Wpk, const float* bias, float* C,
                      int KB, int ldw) {
    int tilesN = ldw / 16;
    int tiles = tilesM * tilesN;
    k_gemm_wmma<<<DIVUP(tiles, 8), 256, 0, stream>>>(Apk, Wpk, bias, C, N, KB,
                                                     ldw, tilesN);
  };

  auto run_edges = [&](int stride, int qoff, int koff, int voff, int H, int C,
                       float scale) {
    int nh = N * H;
    k_fill_f32<<<DIVUP(nh, 256), 256, 0, stream>>>(m, -INFINITY, nh);
    hipMemsetAsync(denom, 0, (size_t)nh * 4, stream);
    hipMemsetAsync(agg, 0, (size_t)N * H * C * 4, stream);
    int eh = E * H;
    int wb = DIVUP(eh, 8);      // wave-per-(e,h) kernels, 8 waves / block
    k_edge_logits<<<wb, 256, 0, stream>>>(Z, src, dst, logits, m, E, H, C, stride,
                                          qoff, koff, scale);
    k_edge_exp<<<DIVUP(eh, 256), 256, 0, stream>>>(logits, dst, m, denom, E, H);
    k_edge_agg<<<wb, 256, 0, stream>>>(Z, src, dst, logits, denom, agg, E, H, C,
                                       stride, voff);
  };

  // ---- layer 1: 13 -> (3 heads x 64), mean heads, relu -------------------
  k_act_pack<<<DIVUP(N * 13, 256), 256, 0, stream>>>(x, Apk, N, 13);
  run_gemm(Wpk1, Bc1, Z, /*KB=*/1, /*ldw=*/640);
  run_edges(640, 0, 192, 384, 3, 64, 0.125f);
  k_out12<<<DIVUP(N * 64, 256), 256, 0, stream>>>(agg, Z, 640, 576, nullptr, h1,
                                                  N, 3, 64);

  // ---- layer 2: 64 -> (3 heads x 64), mean heads, relu, + residual -------
  k_act_pack<<<DIVUP(N * 64, 256), 256, 0, stream>>>(h1, Apk, N, 64);
  run_gemm(Wpk2, Bc2, Z, /*KB=*/2, /*ldw=*/640);
  run_edges(640, 0, 192, 384, 3, 64, 0.125f);
  k_out12<<<DIVUP(N * 64, 256), 256, 0, stream>>>(agg, Z, 640, 576, h1, h2,
                                                  N, 3, 64);

  // ---- layer 3: 64 -> (1 head x 10), concat, log_softmax -----------------
  k_act_pack<<<DIVUP(N * 64, 256), 256, 0, stream>>>(h2, Apk, N, 64);
  run_gemm(Wpk3, Bc3, Z, /*KB=*/2, /*ldw=*/64);
  run_edges(64, 0, 16, 32, 1, 10, 0.31622776601683794f);
  k_out3<<<DIVUP(N, 256), 256, 0, stream>>>(agg, Z, 64, 48, (float*)d_out, N);
}
